// ModelMoESparseBlock_13984413516389
// MI455X (gfx1250) — matrix operand order repaired and checked by reading, not compiled, not run on hardware
//
#include <hip/hip_runtime.h>

#define DEVINL __device__ __forceinline__

typedef __attribute__((ext_vector_type(16))) _Float16 v16h;
typedef __attribute__((ext_vector_type(8)))  _Float16 v8h;
typedef __attribute__((ext_vector_type(8)))  float    v8f;
typedef __attribute__((ext_vector_type(4)))  float    v4f;

constexpr int T_  = 4096;   // tokens
constexpr int HD  = 2048;   // hidden
constexpr int ID  = 1408;   // intermediate
constexpr int NE  = 8;      // experts

constexpr int LSTR = 72;    // LDS row stride in halfs (64 data + 8 pad, rows stay 16B aligned)

// ---------------- workspace layout (bytes) ----------------
constexpr size_t OFF_CNT   = 0;                                   // NE ints
constexpr size_t OFF_RB    = 64;                                  // NE+1 ints
constexpr size_t OFF_EXP   = 256;                                 // 2T ints
constexpr size_t OFF_SLOT  = OFF_EXP  + (size_t)2*T_*4;           // 2T ints
constexpr size_t OFF_WOF   = OFF_SLOT + (size_t)2*T_*4;           // 2T floats
constexpr size_t OFF_TLIST = OFF_WOF  + (size_t)2*T_*4;           // NE*T ints
constexpr size_t OFF_HSH   = OFF_TLIST + (size_t)NE*T_*4;         // T*HD f16
constexpr size_t OFF_W1H   = OFF_HSH  + (size_t)T_*HD*2;          // NE*2I*HD f16
constexpr size_t OFF_W2H   = OFF_W1H  + (size_t)NE*2*ID*HD*2;     // NE*HD*ID f16
constexpr size_t OFF_ACT   = OFF_W2H  + (size_t)NE*HD*ID*2;       // 2T*ID f16
constexpr size_t OFF_OUTR  = OFF_ACT  + (size_t)2*T_*ID*2;        // 2T*HD f32

// ---------------- CDNA5 async global->LDS copy (ASYNCcnt path) ----------
DEVINL void async_ld_b128(_Float16* lds, const _Float16* g) {
  unsigned int l = (unsigned int)(unsigned long long)lds;  // low 32 bits = LDS byte addr
  asm volatile("global_load_async_to_lds_b128 %0, %1, off"
               :: "v"(l), "v"(g) : "memory");
}
DEVINL void wait_async0() {
  asm volatile("s_wait_asynccnt 0x0" ::: "memory");
}

// ---------------- WMMA fragment loaders (LDS, ISA 7.12.2 layouts) -------
DEVINL v16h frag_A(const _Float16* base, int lane) {
  // A 16x32 f16: lane m=lane&15; elems 0..7 -> K=hsel*8.. ; elems 8..15 -> K=16+hsel*8..
  int m = lane & 15, hsel = lane >> 4;
  const _Float16* p = base + m * LSTR;
  v8h lo = *(const v8h*)(p + hsel * 8);
  v8h hi = *(const v8h*)(p + 16 + hsel * 8);
  return __builtin_shufflevector(lo, hi, 0,1,2,3,4,5,6,7,8,9,10,11,12,13,14,15);
}

DEVINL v16h frag_B(const _Float16* base, int lane) {
  // B 32x16: col n=lane&15; elems 0..7 -> K=ksel*8.. ; elems 8..15 -> K=16+ksel*8..
  int n = lane & 15, ksel = lane >> 4;
  const _Float16* p = base + n * LSTR;
  v8h lo = *(const v8h*)(p + ksel * 8);
  v8h hi = *(const v8h*)(p + 16 + ksel * 8);
  return __builtin_shufflevector(lo, hi, 0,1,2,3,4,5,6,7,8,9,10,11,12,13,14,15);
}

// ---------------- 0) f32 -> f16 conversion (one pass, bandwidth bound) ---
__global__ void moe_cvt16(const float* __restrict__ src, _Float16* __restrict__ dst) {
  size_t i = ((size_t)blockIdx.x * blockDim.x + threadIdx.x) * 8;
  v4f a = *(const v4f*)(src + i);
  v4f b = *(const v4f*)(src + i + 4);
  v8h o;
#pragma unroll
  for (int j = 0; j < 4; ++j) { o[j] = (_Float16)a[j]; o[4 + j] = (_Float16)b[j]; }
  *(v8h*)(dst + i) = o;
}

// ---------------- 1) router logits: one wave per token ----------------
__global__ void moe_router(const float* __restrict__ hs, const float* __restrict__ gw,
                           float* __restrict__ logits) {
  int wave = threadIdx.x >> 5, lane = threadIdx.x & 31;
  int t = blockIdx.x * 8 + wave;
  float acc[NE];
#pragma unroll
  for (int e = 0; e < NE; ++e) acc[e] = 0.f;
  const float* hrow = hs + (size_t)t * HD;
  for (int h = lane; h < HD; h += 32) {
    float x = hrow[h];
#pragma unroll
    for (int e = 0; e < NE; ++e) acc[e] += x * gw[e * HD + h];
  }
#pragma unroll
  for (int e = 0; e < NE; ++e) {
#pragma unroll
    for (int off = 16; off > 0; off >>= 1) acc[e] += __shfl_xor(acc[e], off, 32);
  }
  if (lane < NE) logits[(size_t)t * NE + lane] = acc[lane];
}

// ---------------- 2) softmax + top-2 + routing lists ----------------
__global__ void moe_topk(const float* __restrict__ logits, int* __restrict__ counts,
                         int* __restrict__ tlist, int* __restrict__ exp_of,
                         int* __restrict__ slot_of, float* __restrict__ w_of) {
  int t = blockIdx.x * blockDim.x + threadIdx.x;
  if (t >= T_) return;
  float l[NE];
  float mx = -3.4e38f;
#pragma unroll
  for (int e = 0; e < NE; ++e) { l[e] = logits[(size_t)t * NE + e]; mx = fmaxf(mx, l[e]); }
  float p[NE];
#pragma unroll
  for (int e = 0; e < NE; ++e) p[e] = __expf(l[e] - mx);
  int i0 = 0;
#pragma unroll
  for (int e = 1; e < NE; ++e) if (p[e] > p[i0]) i0 = e;
  int i1 = (i0 == 0) ? 1 : 0;
#pragma unroll
  for (int e = 0; e < NE; ++e) if (e != i0 && p[e] > p[i1]) i1 = e;
  float w0 = p[i0], w1 = p[i1];
  float inv = 1.f / (w0 + w1);
  w0 *= inv; w1 *= inv;
  int s0 = atomicAdd(&counts[i0], 1); tlist[(size_t)i0 * T_ + s0] = t;
  int s1 = atomicAdd(&counts[i1], 1); tlist[(size_t)i1 * T_ + s1] = t;
  exp_of[2*t] = i0; slot_of[2*t] = s0; w_of[2*t] = w0;
  exp_of[2*t+1] = i1; slot_of[2*t+1] = s1; w_of[2*t+1] = w1;
}

// ---------------- 3) exclusive scan over E=8 counts ----------------
__global__ void moe_scan(const int* __restrict__ counts, int* __restrict__ row_base) {
  if (threadIdx.x == 0) {
    int acc = 0;
    for (int e = 0; e < NE; ++e) { row_base[e] = acc; acc += counts[e]; }
    row_base[NE] = acc;
  }
}

// ---------------- 4) GEMM1: gathered tokens @ w1[e]^T, fused silu(g)*u ----
// Block tile M=64, N = 64(g)+64(u). Waves: 2x4 grid, each wave 32x32 register
// tile = 4 WMMA per 32-K step, K staged in 64-wide slabs via async-to-LDS.
__global__ __launch_bounds__(256)
void moe_gemm1(const _Float16* __restrict__ hsh, const _Float16* __restrict__ w1h,
               const int* __restrict__ counts, const int* __restrict__ row_base,
               const int* __restrict__ tlist, _Float16* __restrict__ act) {
  __shared__ __attribute__((aligned(32))) _Float16 Alds[64 * LSTR];
  __shared__ __attribute__((aligned(32))) _Float16 Bgld[64 * LSTR];
  __shared__ __attribute__((aligned(32))) _Float16 Buld[64 * LSTR];
  __shared__ float gx[64 * 65];
  __shared__ float ux[64 * 65];
  __shared__ int tokS[64];

  const int e  = blockIdx.z;
  const int n0 = blockIdx.x * 64;       // within [0, ID)
  const int m0 = blockIdx.y * 64;
  const int cnt = counts[e];
  if (m0 >= cnt) return;
  const int rb = row_base[e];
  const int tid = threadIdx.x;
  const int w = tid >> 5, lane = tid & 31;
  const int mw = w & 1;                 // 0..1 : 32-row strip
  const int nw = w >> 1;                // 0..3 : 32-col strip (0,1 -> g ; 2,3 -> u)
  const int ncol = nw & 1;

  if (tid < 64) {
    int idx = m0 + tid;
    if (idx >= cnt) idx = cnt - 1;
    tokS[tid] = tlist[(size_t)e * T_ + idx];
  }
  __syncthreads();

  const _Float16* W1e = w1h + (size_t)e * 2 * ID * HD;
  v8f acc[2][2];
#pragma unroll
  for (int i = 0; i < 2; ++i)
#pragma unroll
    for (int j = 0; j < 2; ++j) acc[i][j] = (v8f)(0.f);

  for (int ks = 0; ks < HD; ks += 64) {
    // async-stage three 64x64 f16 slabs (A gathered, Bg, Bu)
#pragma unroll
    for (int i = 0; i < 2; ++i) {
      int id = tid + i * 256;          // 0..511
      int r = id >> 3, c = (id & 7) * 8;
      async_ld_b128(Alds + r * LSTR + c, hsh + (size_t)tokS[r] * HD + ks + c);
      async_ld_b128(Bgld + r * LSTR + c, W1e + (size_t)(n0 + r) * HD + ks + c);
      async_ld_b128(Buld + r * LSTR + c, W1e + (size_t)(ID + n0 + r) * HD + ks + c);
    }
    wait_async0();
    __syncthreads();

    const _Float16* Bsel = (nw < 2) ? Bgld : Buld;
#pragma unroll
    for (int kk = 0; kk < 64; kk += 32) {
      v16h a0 = frag_A(Alds + (mw * 32 + 0)  * LSTR + kk, lane);
      v16h a1 = frag_A(Alds + (mw * 32 + 16) * LSTR + kk, lane);
      v16h b0 = frag_B(Bsel + (ncol * 32 + 0)  * LSTR + kk, lane);
      v16h b1 = frag_B(Bsel + (ncol * 32 + 16) * LSTR + kk, lane);
      acc[0][0] = __builtin_amdgcn_wmma_f32_16x16x32_f16(false, a0, false, b0, (short)0, acc[0][0], false, false);
      acc[0][1] = __builtin_amdgcn_wmma_f32_16x16x32_f16(false, a0, false, b1, (short)0, acc[0][1], false, false);
      acc[1][0] = __builtin_amdgcn_wmma_f32_16x16x32_f16(false, a1, false, b0, (short)0, acc[1][0], false, false);
      acc[1][1] = __builtin_amdgcn_wmma_f32_16x16x32_f16(false, a1, false, b1, (short)0, acc[1][1], false, false);
    }
    __syncthreads();
  }

  // exchange g/u through LDS, fuse silu(g)*u, emit f16 act rows
  {
    float* ex = (nw < 2) ? gx : ux;
#pragma unroll
    for (int fi = 0; fi < 2; ++fi)
#pragma unroll
      for (int fj = 0; fj < 2; ++fj)
#pragma unroll
        for (int j = 0; j < 8; ++j) {
          int m = mw * 32 + fi * 16 + (lane >> 4) * 8 + j;
          int n = ncol * 32 + fj * 16 + (lane & 15);
          ex[m * 65 + n] = acc[fi][fj][j];
        }
  }
  __syncthreads();
#pragma unroll
  for (int i = 0; i < 16; ++i) {
    int idx = tid + i * 256;
    int m = idx >> 6, n = idx & 63;
    if (m0 + m < cnt) {
      float g = gx[m * 65 + n];
      float u = ux[m * 65 + n];
      float sg = g / (1.f + __expf(-g));
      act[(size_t)(rb + m0 + m) * ID + (n0 + n)] = (_Float16)(sg * u);
    }
  }
}

// ---------------- 5) GEMM2: act @ w2[e]^T -> packed expert rows ----------
// Block tile M=64, N=128; 2x4 wave grid, 4 WMMA per wave per 32-K step.
__global__ __launch_bounds__(256)
void moe_gemm2(const _Float16* __restrict__ act, const _Float16* __restrict__ w2h,
               const int* __restrict__ counts, const int* __restrict__ row_base,
               float* __restrict__ out_rows) {
  __shared__ __attribute__((aligned(32))) _Float16 Alds[64 * LSTR];
  __shared__ __attribute__((aligned(32))) _Float16 Blds[128 * LSTR];

  const int e  = blockIdx.z;
  const int n0 = blockIdx.x * 128;      // within [0, HD)
  const int m0 = blockIdx.y * 64;
  const int cnt = counts[e];
  if (m0 >= cnt) return;
  const int rb = row_base[e];
  const int tid = threadIdx.x;
  const int w = tid >> 5, lane = tid & 31;
  const int mw = w & 1;                 // 32-row strip
  const int nw = w >> 1;                // 0..3 : 32-col strip of the 128 tile

  const _Float16* W2e = w2h + (size_t)e * HD * ID;
  v8f acc[2][2];
#pragma unroll
  for (int i = 0; i < 2; ++i)
#pragma unroll
    for (int j = 0; j < 2; ++j) acc[i][j] = (v8f)(0.f);

  for (int ks = 0; ks < ID; ks += 64) {
    // A slab: 64 rows (packed act rows), B slab: 128 rows of w2[e]
#pragma unroll
    for (int i = 0; i < 2; ++i) {
      int id = tid + i * 256;           // 0..511
      int r = id >> 3, c = (id & 7) * 8;
      int arow = m0 + r; if (arow >= cnt) arow = cnt - 1;
      async_ld_b128(Alds + r * LSTR + c, act + (size_t)(rb + arow) * ID + ks + c);
    }
#pragma unroll
    for (int i = 0; i < 4; ++i) {
      int id = tid + i * 256;           // 0..1023
      int r = id >> 3, c = (id & 7) * 8;
      async_ld_b128(Blds + r * LSTR + c, W2e + (size_t)(n0 + r) * ID + ks + c);
    }
    wait_async0();
    __syncthreads();

#pragma unroll
    for (int kk = 0; kk < 64; kk += 32) {
      v16h a0 = frag_A(Alds + (mw * 32 + 0)  * LSTR + kk, lane);
      v16h a1 = frag_A(Alds + (mw * 32 + 16) * LSTR + kk, lane);
      v16h b0 = frag_B(Blds + (nw * 32 + 0)  * LSTR + kk, lane);
      v16h b1 = frag_B(Blds + (nw * 32 + 16) * LSTR + kk, lane);
      acc[0][0] = __builtin_amdgcn_wmma_f32_16x16x32_f16(false, a0, false, b0, (short)0, acc[0][0], false, false);
      acc[0][1] = __builtin_amdgcn_wmma_f32_16x16x32_f16(false, a0, false, b1, (short)0, acc[0][1], false, false);
      acc[1][0] = __builtin_amdgcn_wmma_f32_16x16x32_f16(false, a1, false, b0, (short)0, acc[1][0], false, false);
      acc[1][1] = __builtin_amdgcn_wmma_f32_16x16x32_f16(false, a1, false, b1, (short)0, acc[1][1], false, false);
    }
    __syncthreads();
  }

#pragma unroll
  for (int fi = 0; fi < 2; ++fi)
#pragma unroll
    for (int fj = 0; fj < 2; ++fj)
#pragma unroll
      for (int j = 0; j < 8; ++j) {
        int m = mw * 32 + fi * 16 + (lane >> 4) * 8 + j;
        int n = nw * 32 + fj * 16 + (lane & 15);
        if (m0 + m < cnt)
          out_rows[(size_t)(rb + m0 + m) * HD + (n0 + n)] = acc[fi][fj][j];
      }
}

// ---------------- 6) combine: final[t] = w0*row0 + w1*row1 ----------------
__global__ void moe_combine(const float* __restrict__ out_rows, const int* __restrict__ exp_of,
                            const int* __restrict__ slot_of, const float* __restrict__ w_of,
                            const int* __restrict__ row_base, float* __restrict__ out) {
  int gid = blockIdx.x * blockDim.x + threadIdx.x;   // T*HD/4 threads
  int t = gid >> 9;            // HD/4 = 512 vec4 per token
  int c = (gid & 511) << 2;
  int e0 = exp_of[2*t], e1 = exp_of[2*t+1];
  size_t r0 = (size_t)(row_base[e0] + slot_of[2*t]);
  size_t r1 = (size_t)(row_base[e1] + slot_of[2*t+1]);
  float w0 = w_of[2*t], w1 = w_of[2*t+1];
  v4f a = *(const v4f*)(out_rows + r0 * HD + c);
  v4f b = *(const v4f*)(out_rows + r1 * HD + c);
  v4f o = w0 * a + w1 * b;
  *(v4f*)(out + (size_t)t * HD + c) = o;
}

// ---------------- launch ----------------
extern "C" void kernel_launch(void* const* d_in, const int* in_sizes, int n_in,
                              void* d_out, int out_size, void* d_ws, size_t ws_size,
                              hipStream_t stream) {
  (void)in_sizes; (void)n_in; (void)out_size; (void)ws_size;
  const float* hs = (const float*)d_in[0];
  const float* gw = (const float*)d_in[1];
  const float* w1 = (const float*)d_in[2];
  const float* w2 = (const float*)d_in[3];
  float* out = (float*)d_out;                      // [T*HD] final, then [T*NE] logits
  float* logits = out + (size_t)T_ * HD;

  char* wsb = (char*)d_ws;
  int*      counts   = (int*)(wsb + OFF_CNT);
  int*      row_base = (int*)(wsb + OFF_RB);
  int*      exp_of   = (int*)(wsb + OFF_EXP);
  int*      slot_of  = (int*)(wsb + OFF_SLOT);
  float*    w_of     = (float*)(wsb + OFF_WOF);
  int*      tlist    = (int*)(wsb + OFF_TLIST);
  _Float16* hsh      = (_Float16*)(wsb + OFF_HSH);
  _Float16* w1h      = (_Float16*)(wsb + OFF_W1H);
  _Float16* w2h      = (_Float16*)(wsb + OFF_W2H);
  _Float16* act      = (_Float16*)(wsb + OFF_ACT);
  float*    out_rows = (float*)(wsb + OFF_OUTR);

  hipMemsetAsync(counts, 0, NE * sizeof(int), stream);

  const size_t n_hs = (size_t)T_ * HD;            // 8,388,608
  const size_t n_w1 = (size_t)NE * 2 * ID * HD;   // 46,137,344
  const size_t n_w2 = (size_t)NE * HD * ID;       // 23,068,672
  moe_cvt16<<<(unsigned)(n_hs / 2048), 256, 0, stream>>>(hs, hsh);
  moe_cvt16<<<(unsigned)(n_w1 / 2048), 256, 0, stream>>>(w1, w1h);
  moe_cvt16<<<(unsigned)(n_w2 / 2048), 256, 0, stream>>>(w2, w2h);

  moe_router<<<T_ / 8, 256, 0, stream>>>(hs, gw, logits);
  moe_topk<<<T_ / 256, 256, 0, stream>>>(logits, counts, tlist, exp_of, slot_of, w_of);
  moe_scan<<<1, 32, 0, stream>>>(counts, row_base);

  dim3 g1(ID / 64, T_ / 64, NE);    // 22 x 64 x 8, blocks past count[e] early-exit
  moe_gemm1<<<g1, 256, 0, stream>>>(hsh, w1h, counts, row_base, tlist, act);

  dim3 g2(HD / 128, T_ / 64, NE);   // 16 x 64 x 8
  moe_gemm2<<<g2, 256, 0, stream>>>(act, w2h, counts, row_base, out_rows);

  moe_combine<<<(T_ * (HD / 4)) / 256, 256, 0, stream>>>(out_rows, exp_of, slot_of, w_of,
                                                         row_base, out);
}
